// TinyChLSTMEncoder_3255585210854
// MI455X (gfx1250) — matrix operand order repaired and checked by reading, not compiled, not run on hardware
//
#include <hip/hip_runtime.h>
#include <math.h>

// Problem sizes (fixed by the reference)
#define Bsz   32
#define Cch   64
#define Tn    1000
#define Hh    16
#define EMBD  32
#define Nseq  (Bsz * Cch)   // 2048 independent sequences

typedef __attribute__((ext_vector_type(16))) _Float16 v16h;
typedef __attribute__((ext_vector_type(2)))  __fp16   v2fp;   // cvt_pkrtz result type
typedef __attribute__((ext_vector_type(8)))  float    v8f;

__device__ __forceinline__ float gelu_exact(float x) {
    return 0.5f * x * (1.0f + erff(x * 0.7071067811865475f));
}

// Native CDNA5 transcendental tanh (single TRANS op, no libm branches).
__device__ __forceinline__ float tanh_hw(float x) {
#if __has_builtin(__builtin_amdgcn_tanhf)
    return __builtin_amdgcn_tanhf(x);
#else
    float y;
    asm("v_tanh_f32 %0, %1\n\tv_nop" : "=v"(y) : "v"(x));
    return y;
#endif
}
// sigmoid via native tanh: 1 mul + 1 TRANS + 1 fma
__device__ __forceinline__ float sigmoid_hw(float x) {
    return __builtin_fmaf(0.5f, tanh_hw(0.5f * x), 0.5f);
}

// pack two f32 into one dword of f16 halves (v_cvt_pk_rtz_f16_f32)
__device__ __forceinline__ unsigned pack2h(float lo, float hi) {
    union { v2fp h2; unsigned u; } pk;
    pk.h2 = __builtin_amdgcn_cvt_pkrtz(lo, hi);
    return pk.u;
}

// ---------------------------------------------------------------------------
// Kernel 1: fused [dwconv5 + BN + GELU] x 2, storing the result TRANSPOSED as
// xT[t * Nseq + row] so the LSTM kernel's per-step loads are fully coalesced.
// One block per (b,c) row; the whole T=1000 row lives in LDS with 2-halo pad.
// ---------------------------------------------------------------------------
__global__ __launch_bounds__(128)
void conv_stack_kernel(const float* __restrict__ x,
                       const float* __restrict__ w1, const float* __restrict__ b1,
                       const float* __restrict__ g1, const float* __restrict__ be1,
                       const float* __restrict__ m1, const float* __restrict__ v1,
                       const float* __restrict__ w2, const float* __restrict__ b2,
                       const float* __restrict__ g2, const float* __restrict__ be2,
                       const float* __restrict__ m2, const float* __restrict__ v2,
                       float* __restrict__ xT)
{
    __shared__ float s0[Tn + 4];
    __shared__ float s1[Tn + 4];

    const int row = blockIdx.x;          // 0..2047
    const int c   = row & (Cch - 1);     // channel
    const int tid = threadIdx.x;

    if (tid < 2) {                       // zero halos (pad=2)
        s0[tid] = 0.f; s0[Tn + 2 + tid] = 0.f;
        s1[tid] = 0.f; s1[Tn + 2 + tid] = 0.f;
    }
    const float* xr = x + (size_t)row * Tn;
    for (int t = tid; t < Tn; t += 128) s0[t + 2] = xr[t];
    __syncthreads();

    float wA[5], wB[5];
#pragma unroll
    for (int k = 0; k < 5; ++k) { wA[k] = w1[c * 5 + k]; wB[k] = w2[c * 5 + k]; }

    // Fold conv-bias into the BN affine: y = conv*inv + (b*inv + beta - mean*inv)
    const float inv1 = g1[c] * rsqrtf(v1[c] + 1e-5f);
    const float add1 = b1[c] * inv1 + be1[c] - m1[c] * inv1;
    const float inv2 = g2[c] * rsqrtf(v2[c] + 1e-5f);
    const float add2 = b2[c] * inv2 + be2[c] - m2[c] * inv2;

    for (int t = tid; t < Tn; t += 128) {
        float a = 0.f;
#pragma unroll
        for (int k = 0; k < 5; ++k) a += s0[t + k] * wA[k];
        s1[t + 2] = gelu_exact(a * inv1 + add1);
    }
    __syncthreads();

    for (int t = tid; t < Tn; t += 128) {
        float a = 0.f;
#pragma unroll
        for (int k = 0; k < 5; ++k) a += s1[t + k] * wB[k];
        xT[(size_t)t * Nseq + row] = gelu_exact(a * inv2 + add2);
    }
}

// ---------------------------------------------------------------------------
// Kernel 2: LSTM over 1000 steps. One wave32 per (16-sequence tile, direction).
//   A (constant in VGPRs) = [Whh gate block | Wih | bias] :
//       K=0..15  -> Whh (recurrence)
//       K=16     -> Wih column (input projection)
//       K=17     -> bih+bhh    (bias)
//   B (rebuilt per step) = [h ; xt ; 1.0] in the matching K slots.
//   => gate pre-activations come straight out of the 4 WMMA accumulators.
//   D = gates[unit][seq]: identical layout for i/f/g/o tiles, c and h, so the
//   nonlinear update is lane-local VALU using native v_tanh_f32 TRANS ops.
//   B rebuild = v_cvt_pk_rtz_f16_f32 packs only; the ds_swizzle result is unused.
// ---------------------------------------------------------------------------
__global__ __launch_bounds__(256)
void lstm_kernel(const float* __restrict__ xT,
                 const float* __restrict__ Wih_f, const float* __restrict__ Whh_f,
                 const float* __restrict__ bih_f, const float* __restrict__ bhh_f,
                 const float* __restrict__ Wih_b, const float* __restrict__ Whh_b,
                 const float* __restrict__ bih_b, const float* __restrict__ bhh_b,
                 float* __restrict__ feat)
{
    const int lane = threadIdx.x & 31;
    const int wid  = blockIdx.x * (blockDim.x >> 5) + (threadIdx.x >> 5); // 0..255
    const int dir  = wid & 1;
    const int tile = wid >> 1;            // 0..127
    const int ml   = lane & 15;
    const int hlf  = lane >> 4;           // which half-wave
    const int ub   = hlf * 8;             // unit / K sub-block base

    const float* Wih = dir ? Wih_b : Wih_f;
    const float* Whh = dir ? Whh_b : Whh_f;
    const float* bih = dir ? bih_b : bih_f;
    const float* bhh = dir ? bhh_b : bhh_f;

    // A tiles (one per gate block, resident for all 1000 steps).
    // 16-bit A 16x32 layout: lanes 0-15 hold K=0..7 (halves 0..7) and K=16..23
    // (halves 8..15) of row M=lane; lanes 16-31 hold K=8..15 and K=24..31.
    // K=16 carries Wih, K=17 carries the bias; all other pad K slots are zero.
    v16h a[4];
#pragma unroll
    for (int G = 0; G < 4; ++G) {
        const int row = G * 16 + ml;      // gate-unit row of this A tile
#pragma unroll
        for (int j = 0; j < 8; ++j) {
            a[G][j]     = (_Float16)Whh[row * 16 + ub + j];
            a[G][8 + j] = (_Float16)0.0f;
        }
        if (hlf == 0) {                   // K=16 (Wih) and K=17 (bias)
            a[G][8] = (_Float16)Wih[row];
            a[G][9] = (_Float16)(bih[row] + bhh[row]);
        }
    }

    const int  seq = tile * 16 + ml;
    const long stp = dir ? -(long)Nseq : (long)Nseq;
    const float* xp = xT + (dir ? (size_t)(Tn - 1) * Nseq : 0) + seq;

    float c8[8], h8[8];
#pragma unroll
    for (int r = 0; r < 8; ++r) { c8[r] = 0.f; h8[r] = 0.f; }

    for (int t = 0; t < Tn; ++t) {
        const float xt = *xp;
        __builtin_prefetch(xp + 8 * stp, 0, 0);     // global_prefetch ahead
        xp += stp;

        // Pack h (f32 -> packed f16) with v_cvt_pk_rtz_f16_f32.
        unsigned own[4];
#pragma unroll
        for (int r = 0; r < 4; ++r) own[r] = pack2h(h8[2 * r], h8[2 * r + 1]);

        // Half-wave exchange (lane ^ 16) via ds_swizzle SWAPX16.
        unsigned oth[4];
#pragma unroll
        for (int r = 0; r < 4; ++r)
            oth[r] = (unsigned)__builtin_amdgcn_ds_swizzle((int)own[r], 0x401F);

        const unsigned xpk = pack2h(xt, 1.0f);      // (K16, K17) = (x, 1)

        // B 32x16 f16 layout: lane n<16 holds K=0..7 (own units 0..7) and
        // K=16..23 (xt, 1.0, 0...) of column n; lane n+16 holds K=8..15
        // (own units 8..15) and K=24..31 (zeros).
        union { unsigned u[8]; v16h v; } Bm;
        Bm.u[0] = own[0];
        Bm.u[1] = own[1];
        Bm.u[2] = own[2];
        Bm.u[3] = own[3];
#pragma unroll
        for (int r = 0; r < 4; ++r) Bm.u[4 + r] = (hlf || r) ? 0u : xpk;

        const v8f z = {};
        v8f gI = __builtin_amdgcn_wmma_f32_16x16x32_f16(false, a[0], false, Bm.v, (short)0, z, false, false);
        v8f gF = __builtin_amdgcn_wmma_f32_16x16x32_f16(false, a[1], false, Bm.v, (short)0, z, false, false);
        v8f gG = __builtin_amdgcn_wmma_f32_16x16x32_f16(false, a[2], false, Bm.v, (short)0, z, false, false);
        v8f gO = __builtin_amdgcn_wmma_f32_16x16x32_f16(false, a[3], false, Bm.v, (short)0, z, false, false);

#pragma unroll
        for (int r = 0; r < 8; ++r) {
            const float i_ = sigmoid_hw(gI[r]);
            const float f_ = sigmoid_hw(gF[r]);
            const float g_ = tanh_hw  (gG[r]);
            const float o_ = sigmoid_hw(gO[r]);
            const float cn = f_ * c8[r] + i_ * g_;
            c8[r] = cn;
            h8[r] = o_ * tanh_hw(cn);
        }
    }

    // Final hidden -> feat[seq][dir*16 + unit]; lane owns units ub..ub+7 of seq.
    float* fr = feat + (size_t)seq * (2 * Hh) + dir * Hh + ub;
#pragma unroll
    for (int r = 0; r < 8; ++r) fr[r] = h8[r];
}

// ---------------------------------------------------------------------------
// Kernel 3: out = feat @ Wlin^T + blin via WMMA. One wave per 16-seq tile.
// K=32 matches v_wmma_f32_16x16x32_f16 exactly; EMB=32 -> two N-tiles.
// ---------------------------------------------------------------------------
__global__ __launch_bounds__(256)
void linear_kernel(const float* __restrict__ feat,
                   const float* __restrict__ Wlin,
                   const float* __restrict__ blin,
                   float* __restrict__ out)
{
    const int lane = threadIdx.x & 31;
    const int tile = blockIdx.x * (blockDim.x >> 5) + (threadIdx.x >> 5); // 0..127
    const int ml   = lane & 15;
    const int hlf  = lane >> 4;
    const int kofs = hlf * 16;

    // A = feat tile: M=16 seqs, K=32 feature dims.
    v16h A;
    const float* fr = feat + (size_t)(tile * 16 + ml) * (2 * Hh) + kofs;
#pragma unroll
    for (int j = 0; j < 16; ++j) A[j] = (_Float16)fr[j];

#pragma unroll
    for (int nb = 0; nb < 2; ++nb) {
        // B = Wlin^T column block: lane n holds K=kofs..kofs+15 of emb column e.
        const int e = nb * 16 + ml;
        const float* wr = Wlin + (size_t)e * (2 * Hh) + kofs;
        v16h Bv;
#pragma unroll
        for (int j = 0; j < 16; ++j) Bv[j] = (_Float16)wr[j];

        v8f acc = {};
        acc = __builtin_amdgcn_wmma_f32_16x16x32_f16(false, A, false, Bv, (short)0, acc, false, false);

        const float bl = blin[e];
#pragma unroll
        for (int r = 0; r < 8; ++r) {
            const int s = tile * 16 + r + hlf * 8;   // D row (sequence)
            out[(size_t)s * EMBD + e] = acc[r] + bl;
        }
    }
}

// ---------------------------------------------------------------------------
extern "C" void kernel_launch(void* const* d_in, const int* in_sizes, int n_in,
                              void* d_out, int out_size, void* d_ws, size_t ws_size,
                              hipStream_t stream)
{
    const float* x     = (const float*)d_in[0];
    const float* w1    = (const float*)d_in[1];
    const float* b1    = (const float*)d_in[2];
    const float* g1    = (const float*)d_in[3];
    const float* be1   = (const float*)d_in[4];
    const float* m1    = (const float*)d_in[5];
    const float* v1    = (const float*)d_in[6];
    const float* w2    = (const float*)d_in[7];
    const float* b2    = (const float*)d_in[8];
    const float* g2    = (const float*)d_in[9];
    const float* be2   = (const float*)d_in[10];
    const float* m2    = (const float*)d_in[11];
    const float* v2    = (const float*)d_in[12];
    const float* Wih_f = (const float*)d_in[13];
    const float* Whh_f = (const float*)d_in[14];
    const float* bih_f = (const float*)d_in[15];
    const float* bhh_f = (const float*)d_in[16];
    const float* Wih_b = (const float*)d_in[17];
    const float* Whh_b = (const float*)d_in[18];
    const float* bih_b = (const float*)d_in[19];
    const float* bhh_b = (const float*)d_in[20];
    const float* Wlin  = (const float*)d_in[21];
    const float* blin  = (const float*)d_in[22];

    float* xT   = (float*)d_ws;                       // (T, Nseq) = 8.19 MB
    float* feat = xT + (size_t)Tn * Nseq;             // (Nseq, 2H) = 256 KB
    float* out  = (float*)d_out;

    conv_stack_kernel<<<Nseq, 128, 0, stream>>>(x, w1, b1, g1, be1, m1, v1,
                                                w2, b2, g2, be2, m2, v2, xT);

    // 256 waves of LSTM work (128 tiles x 2 directions), 8 waves/block.
    lstm_kernel<<<32, 256, 0, stream>>>(xT, Wih_f, Whh_f, bih_f, bhh_f,
                                        Wih_b, Whh_b, bih_b, bhh_b, feat);

    // 128 waves of linear work, 8 waves/block.
    linear_kernel<<<16, 256, 0, stream>>>(feat, Wlin, blin, out);
}
